// SynthesisBlock_91268055040377
// MI455X (gfx1250) — matrix operand order repaired and checked by reading, not compiled, not run on hardware
//
#include <hip/hip_runtime.h>
#include <hip/hip_bf16.h>

typedef _Float16 h8  __attribute__((ext_vector_type(8)));
typedef _Float16 h16 __attribute__((ext_vector_type(16)));
typedef float    f8  __attribute__((ext_vector_type(8)));

#define NB   8
#define H0   64
#define W0   64
#define CI0  512
#define CO   256
#define H1   128
#define W1   128
#define HP0  66          // padded 64+2
#define HP1  130         // padded 128+2

// padded buffer element counts (halves)
#define X0F_HALVES  ((size_t)NB * HP0 * HP0 * CI0)   // 17,842,176
#define X1M_HALVES  ((size_t)NB * HP1 * HP1 * CO)    // 34,611,200

__device__ __forceinline__ float wave_reduce(float v) {
#pragma unroll
    for (int off = 16; off; off >>= 1) v += __shfl_down(v, off, 32);
    return v;
}

// A fragment: two contiguous 16B halves (ISA 16-bit A layout: grp g -> K {8g..}U{16+8g..})
__device__ __forceinline__ h16 load_a_frag(const _Float16* p) {
    h8 lo = *(const h8*)p;
    h8 hi = *(const h8*)(p + 16);
    return __builtin_shufflevector(lo, hi, 0,1,2,3,4,5,6,7,8,9,10,11,12,13,14,15);
}
__device__ __forceinline__ h16 load_b_frag(const _Float16* p) {
    return *(const h16*)p;    // 32B contiguous, pre-packed in B layout
}

// ---------------------------- zero fill (halo init) --------------------------
__global__ void fill0_kernel(_Float16* __restrict__ p) {
    size_t t = (size_t)blockIdx.x * blockDim.x + threadIdx.x;
    *(h8*)(p + t * 8) = (h8){};
}

// ---------------- affine: s = dl[:,layer] @ (w/sqrt(512)) + b ----------------
__global__ void affine_kernel(const float* __restrict__ dl,
                              const float* __restrict__ w,
                              const float* __restrict__ b,
                              float* __restrict__ out,
                              int layer, int n_out) {
    int gw   = (blockIdx.x * blockDim.x + threadIdx.x) >> 5;
    int lane = threadIdx.x & 31;
    int n = gw / n_out, o = gw - n * n_out;
    const float* v = dl + ((size_t)n * 12 + layer) * 512;
    float acc = 0.f;
    for (int i = lane; i < 512; i += 32) acc += v[i] * w[(size_t)i * n_out + o];
    acc = wave_reduce(acc);
    if (!lane) out[n * n_out + o] = acc * 0.04419417382415922f /*1/sqrt(512)*/ + b[o];
}

// ------------- demod: d[n,co] = rsqrt(sum (w_eq * s)^2 + 1e-8) ---------------
__global__ void demod_kernel(const float* __restrict__ w,
                             const float* __restrict__ s,
                             float* __restrict__ d,
                             int Ci, int Co_, float eq) {
    int gw   = (blockIdx.x * blockDim.x + threadIdx.x) >> 5;
    int lane = threadIdx.x & 31;
    int n = gw / Co_, co = gw - n * Co_;
    float acc = 0.f;
    for (int t = 0; t < 9; ++t) {
        const float* wt = w + (size_t)t * Ci * Co_;
        for (int ci = lane; ci < Ci; ci += 32) {
            float v = wt[(size_t)ci * Co_ + co] * eq * s[n * Ci + ci];
            acc += v * v;
        }
    }
    acc = wave_reduce(acc);
    if (!lane) d[n * Co_ + co] = rsqrtf(acc + 1e-8f);
}

// ---- modulate input by s0, cast f16, write padded interior [8][66][66][512] --
__global__ void modcast_kernel(const float* __restrict__ x,
                               const float* __restrict__ s,
                               _Float16* __restrict__ o) {
    size_t t = (size_t)blockIdx.x * blockDim.x + threadIdx.x;
    int c  = (int)(t & (CI0 - 1));
    int r  = (int)(t >> 9);
    int wx = r & 63; r >>= 6;
    int wy = r & 63;
    int n  = r >> 6;
    o[((size_t)(n * HP0 + wy + 1) * HP0 + wx + 1) * CI0 + c] = (_Float16)(x[t] * s[n * CI0 + c]);
}

// ---- pack layer0 composite (convT(3x3,s2) ∘ FIR4x4*4) into B-fragments ------
// parity (py,px): out[2a+py,2b+px] = sum_{dj in 3x3} g2[2dj+3-p] . x[a+dj]
// g2[ry,rx] = (1/16) * sum_{ty,tx} k[3-(ry-ty)] k[3-(rx-tx)] w[2-ty,2-tx]
// B layout (16x16x32 f16 WMMA): lane L -> col N=L&15, grp=L>>4; K(e) = 8*grp+(e&7)+16*(e>>3)
__global__ void packw0_kernel(const float* __restrict__ w, _Float16* __restrict__ o) {
    int t = blockIdx.x * blockDim.x + threadIdx.x;      // 4,718,592 total
    int e = t & 15, lane = (t >> 4) & 31, kc = (t >> 9) & 15, cot = (t >> 13) & 15;
    int rem = t >> 17; int tap = rem % 9; int p = rem / 9;
    int g  = lane >> 4;
    int co = cot * 16 + (lane & 15);
    int ci = kc * 32 + 8 * g + (e & 7) + ((e >> 3) << 4);
    int py = p >> 1, px = p & 1;
    int tapy = tap / 3, tapx = tap % 3;
    int ry = 2 * tapy + 1 - py, rx = 2 * tapx + 1 - px;
    const float kf[4] = {1.f, 3.f, 3.f, 1.f};
    float acc = 0.f;
    for (int ty = 0; ty < 3; ++ty) {
        int uy = ry - ty; if (uy < 0 || uy > 3) continue;
        for (int tx = 0; tx < 3; ++tx) {
            int ux = rx - tx; if (ux < 0 || ux > 3) continue;
            acc += kf[3 - uy] * kf[3 - ux] *
                   w[(size_t)(((2 - ty) * 3 + (2 - tx)) * CI0 + ci) * CO + co];
        }
    }
    // (1/16) FIR factor * eq scale 1/sqrt(9*512)
    o[t] = (_Float16)(acc * 0.0625f * 0.014731391274719738f);
}

// ------------- pack layer1 3x3 weights (eq = 1/48) into B-fragments ----------
__global__ void packw1_kernel(const float* __restrict__ w, _Float16* __restrict__ o) {
    int t = blockIdx.x * blockDim.x + threadIdx.x;      // 589,824 total
    int e = t & 15, lane = (t >> 4) & 31, kc = (t >> 9) & 7, cot = (t >> 12) & 15, tap = t >> 16;
    int g  = lane >> 4;
    int co = cot * 16 + (lane & 15);
    int ci = kc * 32 + 8 * g + (e & 7) + ((e >> 3) << 4);
    o[t] = (_Float16)(w[(size_t)(tap * CO + ci) * CO + co] * 0.020833333333333332f);
}

// -------- layer0: parity-decomposed 3x3 WMMA conv over 64x64, 512->256 -------
// block = 8 waves sharing (n,p,cog): wave tile M=16 px, N=64 co; B identical
// across all 8 waves (WGP-cache reuse); A loads unconditional via zero halo;
// 2-stage software pipeline (fully unrolled) so loads overlap the WMMA pipe.
__global__ void conv_up_kernel(const _Float16* __restrict__ x0,
                               const _Float16* __restrict__ w0p,
                               const float* __restrict__ d0,
                               const float* __restrict__ s1,
                               const float* __restrict__ bias,
                               const float* __restrict__ noise,
                               const float* __restrict__ ns,
                               _Float16* __restrict__ x1m) {
    int wv   = threadIdx.x >> 5;
    int lane = threadIdx.x & 31;
    int axt = wv & 3, ayl = wv >> 2;
    int b = blockIdx.x;
    int cog = b & 3, ayh = (b >> 2) & 31, p = (b >> 7) & 3, n = b >> 9;
    int ay = ayh * 2 + ayl;
    int py = p >> 1, px = p & 1;
    int ax0 = axt * 16;
    int g = lane >> 4, col = lane & 15;
    f8 acc[4] = {};
    float nsv = ns[0];

    // padded A base for tap(0,0): padded coords (ay-1+1, ax0+col-1+1)
    const _Float16* abase = x0 + ((size_t)(n * HP0 + ay) * HP0 + ax0 + col) * CI0 + 8 * g;

#pragma unroll
    for (int tapy = 0; tapy < 3; ++tapy) {
#pragma unroll
        for (int tapx = 0; tapx < 3; ++tapx) {
            const _Float16* ap0 = abase + (size_t)(tapy * HP0 + tapx) * CI0;
            int tap = tapy * 3 + tapx;
            const _Float16* bp0 = w0p + ((size_t)((p * 9 + tap) * 16 + cog * 4)) * 8192
                                      + (size_t)lane * 16;
            h16 aa[2], bb[2][4];
            aa[0]    = load_a_frag(ap0);
            bb[0][0] = load_b_frag(bp0);
            bb[0][1] = load_b_frag(bp0 + 8192);
            bb[0][2] = load_b_frag(bp0 + 16384);
            bb[0][3] = load_b_frag(bp0 + 24576);
#pragma unroll
            for (int kc = 0; kc < 16; ++kc) {
                int cur = kc & 1, nxt = cur ^ 1;
                if (kc + 1 < 16) {   // prefetch next K-slab into fresh registers
                    const _Float16* ap = ap0 + (kc + 1) * 32;
                    const _Float16* bp = bp0 + (size_t)(kc + 1) * 512;
                    aa[nxt]    = load_a_frag(ap);
                    bb[nxt][0] = load_b_frag(bp);
                    bb[nxt][1] = load_b_frag(bp + 8192);
                    bb[nxt][2] = load_b_frag(bp + 16384);
                    bb[nxt][3] = load_b_frag(bp + 24576);
                }
                acc[0] = __builtin_amdgcn_wmma_f32_16x16x32_f16(false, aa[cur], false, bb[cur][0], (short)0, acc[0], false, false);
                acc[1] = __builtin_amdgcn_wmma_f32_16x16x32_f16(false, aa[cur], false, bb[cur][1], (short)0, acc[1], false, false);
                acc[2] = __builtin_amdgcn_wmma_f32_16x16x32_f16(false, aa[cur], false, bb[cur][2], (short)0, acc[2], false, false);
                acc[3] = __builtin_amdgcn_wmma_f32_16x16x32_f16(false, aa[cur], false, bb[cur][3], (short)0, acc[3], false, false);
            }
        }
    }
    // epilogue: demod + noise + bias + lrelu*sqrt2, then fold s1 + f16 cast
    int oy = 2 * ay + py;
#pragma unroll
    for (int i = 0; i < 4; ++i) {
        int co = cog * 64 + i * 16 + col;
        float dm = d0[n * CO + co], bi = bias[co], sm = s1[n * CO + co];
#pragma unroll
        for (int r = 0; r < 8; ++r) {
            int m  = r + 8 * g;                 // D layout: M = r + 8*grp, N = lane&15
            int ox = 2 * (ax0 + m) + px;
            float v = acc[i][r] * dm + noise[oy * W1 + ox] * nsv + bi;
            v = (v > 0.f ? v : 0.2f * v) * 1.4142135623730951f;
            x1m[((size_t)(n * HP1 + oy + 1) * HP1 + ox + 1) * CO + co] = (_Float16)(v * sm);
        }
    }
}

// --------------- layer1: 3x3 SAME WMMA conv over 128x128, 256->256 -----------
// block = 8 waves (8 ox tiles) sharing (n,oy,cog): B identical across block.
__global__ void conv3x3_kernel(const _Float16* __restrict__ x1m,
                               const _Float16* __restrict__ w1p,
                               const float* __restrict__ d1,
                               const float* __restrict__ bias,
                               const float* __restrict__ noise,
                               const float* __restrict__ ns,
                               float* __restrict__ xout) {
    int oxt  = threadIdx.x >> 5;
    int lane = threadIdx.x & 31;
    int b = blockIdx.x;
    int cog = b & 3, oy = (b >> 2) & 127, n = b >> 9;
    int ox0 = oxt * 16;
    int g = lane >> 4, col = lane & 15;
    f8 acc[4] = {};
    float nsv = ns[0];

    const _Float16* abase = x1m + ((size_t)(n * HP1 + oy) * HP1 + ox0 + col) * CO + 8 * g;

#pragma unroll
    for (int tapy = 0; tapy < 3; ++tapy) {
#pragma unroll
        for (int tapx = 0; tapx < 3; ++tapx) {
            const _Float16* ap0 = abase + (size_t)(tapy * HP1 + tapx) * CO;
            int tap = tapy * 3 + tapx;
            const _Float16* bp0 = w1p + ((size_t)(tap * 16 + cog * 4)) * 4096
                                      + (size_t)lane * 16;
            h16 aa[2], bb[2][4];
            aa[0]    = load_a_frag(ap0);
            bb[0][0] = load_b_frag(bp0);
            bb[0][1] = load_b_frag(bp0 + 4096);
            bb[0][2] = load_b_frag(bp0 + 8192);
            bb[0][3] = load_b_frag(bp0 + 12288);
#pragma unroll
            for (int kc = 0; kc < 8; ++kc) {
                int cur = kc & 1, nxt = cur ^ 1;
                if (kc + 1 < 8) {
                    const _Float16* ap = ap0 + (kc + 1) * 32;
                    const _Float16* bp = bp0 + (size_t)(kc + 1) * 512;
                    aa[nxt]    = load_a_frag(ap);
                    bb[nxt][0] = load_b_frag(bp);
                    bb[nxt][1] = load_b_frag(bp + 4096);
                    bb[nxt][2] = load_b_frag(bp + 8192);
                    bb[nxt][3] = load_b_frag(bp + 12288);
                }
                acc[0] = __builtin_amdgcn_wmma_f32_16x16x32_f16(false, aa[cur], false, bb[cur][0], (short)0, acc[0], false, false);
                acc[1] = __builtin_amdgcn_wmma_f32_16x16x32_f16(false, aa[cur], false, bb[cur][1], (short)0, acc[1], false, false);
                acc[2] = __builtin_amdgcn_wmma_f32_16x16x32_f16(false, aa[cur], false, bb[cur][2], (short)0, acc[2], false, false);
                acc[3] = __builtin_amdgcn_wmma_f32_16x16x32_f16(false, aa[cur], false, bb[cur][3], (short)0, acc[3], false, false);
            }
        }
    }
#pragma unroll
    for (int i = 0; i < 4; ++i) {
        int co = cog * 64 + i * 16 + col;
        float dm = d1[n * CO + co], bi = bias[co];
#pragma unroll
        for (int r = 0; r < 8; ++r) {
            int ox = ox0 + r + 8 * g;
            float v = acc[i][r] * dm + noise[oy * W1 + ox] * nsv + bi;
            v = (v > 0.f ? v : 0.2f * v) * 1.4142135623730951f;
            xout[((size_t)((n * H1 + oy) * W1 + ox)) * CO + co] = v;
        }
    }
}

// ------ ToRGB: 1x1 conv 256->3 (mod by srgb, eq=1/16) + bias + upsample(y) ---
__global__ void torgb_kernel(const float* __restrict__ xout,
                             const float* __restrict__ srgb,
                             const float* __restrict__ rgbw,
                             const float* __restrict__ rgbb,
                             const float* __restrict__ y,
                             float* __restrict__ img) {
    int gw   = blockIdx.x * 8 + (threadIdx.x >> 5);
    int lane = threadIdx.x & 31;
    int ox = gw & 127, oy = (gw >> 7) & 127, n = gw >> 14;
    const float* xr = xout + ((size_t)((n * H1 + oy) * W1 + ox)) * CO;
    const float* sv = srgb + n * CO;
    float r = 0.f, gc = 0.f, bc = 0.f;
#pragma unroll
    for (int t = 0; t < 8; ++t) {
        int ci = lane + 32 * t;
        float xv = xr[ci] * sv[ci] * 0.0625f;     // eq = 1/sqrt(256)
        r  += xv * rgbw[ci * 3 + 0];
        gc += xv * rgbw[ci * 3 + 1];
        bc += xv * rgbw[ci * 3 + 2];
    }
    r = wave_reduce(r); gc = wave_reduce(gc); bc = wave_reduce(bc);
    if (!lane) {
        const float kf[4] = {1.f, 3.f, 3.f, 1.f};
        float up0 = 0.f, up1 = 0.f, up2 = 0.f;
        for (int ty = 0; ty < 4; ++ty) {
            int dy = oy + ty - 2; if (dy < 0 || (dy & 1) || dy > 126) continue;
            int iy = dy >> 1; float wy = kf[3 - ty];
            for (int tx = 0; tx < 4; ++tx) {
                int dx = ox + tx - 2; if (dx < 0 || (dx & 1) || dx > 126) continue;
                int ix = dx >> 1;
                float wgt = wy * kf[3 - tx] * 0.0625f;   // gain4 * f2 = k[a]k[b]/16
                const float* yp = y + ((size_t)((n * H0 + iy) * W0 + ix)) * 3;
                up0 += wgt * yp[0]; up1 += wgt * yp[1]; up2 += wgt * yp[2];
            }
        }
        float* op = img + ((size_t)((n * H1 + oy) * W1 + ox)) * 3;
        op[0] = r + rgbb[0] + up0;
        op[1] = gc + rgbb[1] + up1;
        op[2] = bc + rgbb[2] + up2;
    }
}

extern "C" void kernel_launch(void* const* d_in, const int* in_sizes, int n_in,
                              void* d_out, int out_size, void* d_ws, size_t ws_size,
                              hipStream_t stream) {
    (void)in_sizes; (void)n_in; (void)out_size; (void)ws_size;
    const float* x       = (const float*)d_in[0];
    const float* y       = (const float*)d_in[1];
    const float* dl      = (const float*)d_in[2];
    const float* noise0  = (const float*)d_in[3];
    const float* noise1  = (const float*)d_in[4];
    const float* aff0_w  = (const float*)d_in[5];
    const float* aff0_b  = (const float*)d_in[6];
    const float* conv0_w = (const float*)d_in[7];
    const float* conv0_b = (const float*)d_in[8];
    const float* ns0     = (const float*)d_in[9];
    const float* aff1_w  = (const float*)d_in[10];
    const float* aff1_b  = (const float*)d_in[11];
    const float* conv1_w = (const float*)d_in[12];
    const float* conv1_b = (const float*)d_in[13];
    const float* ns1     = (const float*)d_in[14];
    const float* rgb_aw  = (const float*)d_in[15];
    const float* rgb_ab  = (const float*)d_in[16];
    const float* rgb_w   = (const float*)d_in[17];
    const float* rgb_b   = (const float*)d_in[18];

    char* ws = (char*)d_ws;
    size_t off = 0;
    auto take = [&](size_t bytes) -> char* {
        char* p = ws + off;
        off += (bytes + 255) & ~(size_t)255;
        return p;
    };
    float*    s0   = (float*)take(8 * 512 * 4);
    float*    s1   = (float*)take(8 * 256 * 4);
    float*    srgb = (float*)take(8 * 256 * 4);
    float*    d0   = (float*)take(8 * 256 * 4);
    float*    d1   = (float*)take(8 * 256 * 4);
    _Float16* w0p  = (_Float16*)take((size_t)4718592 * 2);   // 4par*9tap*512K*256N packed
    _Float16* w1p  = (_Float16*)take((size_t)589824 * 2);    // 9tap*256K*256N packed
    _Float16* x0f  = (_Float16*)take(X0F_HALVES * 2);        // padded modulated input f16
    _Float16* x1m  = (_Float16*)take(X1M_HALVES * 2);        // padded layer1 input f16

    float* xout = (float*)d_out;
    float* img  = xout + (size_t)8 * 128 * 128 * 256;

    // zero-fill padded activation buffers (halo must be 0)
    fill0_kernel<<<(unsigned)(X0F_HALVES / 8 / 256), 256, 0, stream>>>(x0f);
    fill0_kernel<<<(unsigned)(X1M_HALVES / 8 / 256), 256, 0, stream>>>(x1m);

    affine_kernel<<<512, 256, 0, stream>>>(dl, aff0_w, aff0_b, s0, 9, 512);
    affine_kernel<<<256, 256, 0, stream>>>(dl, aff1_w, aff1_b, s1, 10, 256);
    affine_kernel<<<256, 256, 0, stream>>>(dl, rgb_aw, rgb_ab, srgb, 11, 256);
    demod_kernel<<<256, 256, 0, stream>>>(conv0_w, s0, d0, 512, 256, 0.014731391274719738f);
    demod_kernel<<<256, 256, 0, stream>>>(conv1_w, s1, d1, 256, 256, 0.020833333333333332f);
    modcast_kernel<<<65536, 256, 0, stream>>>(x, s0, x0f);
    packw0_kernel<<<18432, 256, 0, stream>>>(conv0_w, w0p);
    packw1_kernel<<<2304, 256, 0, stream>>>(conv1_w, w1p);
    conv_up_kernel<<<4096, 256, 0, stream>>>(x0f, w0p, d0, s1, conv0_b, noise0, ns0, x1m);
    conv3x3_kernel<<<4096, 256, 0, stream>>>(x1m, w1p, d1, conv1_b, noise1, ns1, xout);
    torgb_kernel<<<16384, 256, 0, stream>>>(xout, srgb, rgb_w, rgb_b, y, img);
}
